// biGATNet_67259187855851
// MI455X (gfx1250) — compile-verified, matrix operand order checked by reading, not run on hardware
//
#include <hip/hip_runtime.h>
#include <hip/hip_bf16.h>
#include <math.h>

#define NN 40000
#define EE 640000
#define GRID(n) (((n) + 255) / 256)

typedef __attribute__((ext_vector_type(16))) _Float16 v16h;
typedef __attribute__((ext_vector_type(8)))  float    v8f;

// ---------------- gfx1250 async global->LDS staging ----------------
// GLOBAL_LOAD_ASYNC_TO_LDS_B128: VDST = VGPR holding LDS byte address,
// VADDR = 64-bit global address (GV mode). Tracked by ASYNCcnt.
static __device__ inline void async_ld16(unsigned lds_off, const void* gp) {
    unsigned long long ga = (unsigned long long)(uintptr_t)gp;
    asm volatile("global_load_async_to_lds_b128 %0, %1, off"
                 :: "v"(lds_off), "v"(ga) : "memory");
}
static __device__ inline void wait_async0() {
    asm volatile("s_wait_asynccnt 0x0" ::: "memory");
}

// ---------------- WMMA fragment loaders (wave32, 16x16x32 f16) ----------------
// A fragment: 16 rows x 32 K. lanes 0-15: rows 0-15, K halves {0-7,16-23};
// lanes 16-31: rows 0-15, K halves {8-15,24-31}. (ISA 7.12.2 16-bit A layout)
static __device__ inline v16h load_a_frag(const _Float16* Arow0, int ld, int k0, int lane) {
    int m  = lane & 15;
    int kh = (lane >> 4) & 1;
    const _Float16* p = Arow0 + (size_t)m * ld + k0;
    v16h a;
#pragma unroll
    for (int j = 0; j < 8; ++j) a[j] = p[kh * 8 + j];
#pragma unroll
    for (int j = 0; j < 8; ++j) a[8 + j] = p[16 + kh * 8 + j];
    return a;
}

// ---------------- generic WMMA GEMM, K=128: C[M,Ncol] = A[M,128] @ W (+bias,relu) ----
// A tile (16x128 f16, 4KB) staged global->LDS via async-DMA once per block.
// W pre-packed in fragment order: frag(kt,t) = 512 halves, lane-major.
__global__ __launch_bounds__(256) void k_gemm(const _Float16* __restrict__ A,
                                              const _Float16* __restrict__ Wp,
                                              const float* __restrict__ bias, int bias_n,
                                              float* __restrict__ C,
                                              int Ncol, int relu) {
    __shared__ __align__(16) _Float16 atile[16 * 128];
    int row0 = blockIdx.x * 16;
    int tid  = threadIdx.x;
    {   // 256 threads x 16B = full 4KB tile
        int row = tid >> 4, hcol = (tid & 15) << 3;
        async_ld16((unsigned)(uintptr_t)(atile + (tid << 3)),
                   A + (size_t)(row0 + row) * 128 + hcol);
        wait_async0();
    }
    __syncthreads();

    int wave = tid >> 5, lane = tid & 31;
    int nt = Ncol >> 4;
    for (int t = wave; t < nt; t += 8) {
        v8f acc = {};
#pragma unroll
        for (int kt = 0; kt < 4; ++kt) {
            v16h af = load_a_frag(atile, 128, kt * 32, lane);
            const v16h* bp = (const v16h*)(Wp + (size_t)((kt * nt + t) << 9));
            v16h bf = bp[lane];
            acc = __builtin_amdgcn_wmma_f32_16x16x32_f16(false, af, false, bf,
                                                         (short)0, acc, false, false);
        }
        int col0 = t * 16;
        int n = lane & 15, mb = (lane >> 4) * 8;
        float bv = (bias && (col0 + n) < bias_n) ? bias[col0 + n] : 0.f;
#pragma unroll
        for (int r = 0; r < 8; ++r) {
            float v = acc[r] + bv;
            if (relu) v = v > 0.f ? v : 0.f;
            C[(size_t)(row0 + mb + r) * Ncol + col0 + n] = v;
        }
    }
}

// ---------------- conversions / weight fragment packing ----------------
__global__ void k_f32_to_f16(const float* __restrict__ in, _Float16* __restrict__ out, int n) {
    int i = blockIdx.x * blockDim.x + threadIdx.x;
    if (i < n) out[i] = (_Float16)in[i];
}

// Pack W[K, realN] (row-major f32) into per-fragment lane-major f16 layout.
// out index = ((kt*nt + t)*32 + lane)*16 + j ; cols >= realN zero-padded.
__global__ void k_pack(const float* __restrict__ W, _Float16* __restrict__ out,
                       int K, int Ncol, int realN) {
    int i = blockIdx.x * blockDim.x + threadIdx.x;
    if (i >= K * Ncol) return;
    int frag = i >> 9;
    int r    = i & 511;
    int lane = r >> 4, j = r & 15;
    int nt = Ncol >> 4;
    int kt = frag / nt, t = frag % nt;
    int kh = lane >> 4, n = lane & 15;
    int koff = (j < 8) ? (kh * 8 + j) : (16 + kh * 8 + (j - 8));
    int k   = kt * 32 + koff;
    int col = t * 16 + n;
    out[i] = (col < realN) ? (_Float16)W[(size_t)k * realN + col] : (_Float16)0.f;
}

// ---------------- attention pipeline ----------------
static __device__ inline unsigned fkey(float x) {
    unsigned b = __float_as_uint(x);
    return (b & 0x80000000u) ? ~b : (b | 0x80000000u);
}
static __device__ inline float funkey(unsigned k) {
    unsigned b = (k & 0x80000000u) ? (k ^ 0x80000000u) : ~k;
    return __uint_as_float(b);
}

__global__ void k_el_er(const float* __restrict__ feat, const float* __restrict__ al,
                        const float* __restrict__ ar, float* __restrict__ el,
                        float* __restrict__ er, int H, int D) {
    int i = blockIdx.x * blockDim.x + threadIdx.x;
    if (i >= NN * H) return;
    int n = i / H, h = i % H;
    const float* f  = feat + (size_t)n * 128 + h * D;
    const float* a1 = al + h * D;
    const float* a2 = ar + h * D;
    float s1 = 0.f, s2 = 0.f;
    for (int d = 0; d < D; ++d) { s1 += f[d] * a1[d]; s2 += f[d] * a2[d]; }
    el[i] = s1; er[i] = s2;
}

__global__ void k_edge_max(const float* __restrict__ el, const float* __restrict__ er,
                           const int* __restrict__ src, const int* __restrict__ dst,
                           unsigned* __restrict__ mkey, int H) {
    int i = blockIdx.x * blockDim.x + threadIdx.x;
    if (i >= EE * H) return;
    int e = i / H, h = i % H;
    float lg = el[src[e] * H + h] + er[dst[e] * H + h];
    lg = lg > 0.f ? lg : 0.2f * lg;
    atomicMax(&mkey[dst[e] * H + h], fkey(lg));
}

__global__ void k_edge_exp(const float* __restrict__ el, const float* __restrict__ er,
                           const int* __restrict__ src, const int* __restrict__ dst,
                           const unsigned* __restrict__ mkey, float* __restrict__ den,
                           float* __restrict__ ex, int H) {
    int i = blockIdx.x * blockDim.x + threadIdx.x;
    if (i >= EE * H) return;
    int e = i / H, h = i % H;
    float lg = el[src[e] * H + h] + er[dst[e] * H + h];
    lg = lg > 0.f ? lg : 0.2f * lg;
    float m = funkey(mkey[dst[e] * H + h]);
    float v = __expf(lg - m);
    ex[i] = v;
    atomicAdd(&den[dst[e] * H + h], v);
}

__global__ void k_scatter(const float* __restrict__ feat, const float* __restrict__ ex,
                          const float* __restrict__ den, const int* __restrict__ src,
                          const int* __restrict__ dst, float* __restrict__ out,
                          int H, int Dshift) {
    int i = blockIdx.x * blockDim.x + threadIdx.x;   // EE*128 threads
    int e = i >> 7, c = i & 127;
    int h = c >> Dshift;
    float w = ex[e * H + h] / den[dst[e] * H + h];
    atomicAdd(&out[(size_t)dst[e] * 128 + c], w * feat[(size_t)src[e] * 128 + c]);
}

// ---------------- batchnorm + ELU + residual + inf-mask ----------------
__global__ __launch_bounds__(256) void k_bn_stats(const float* __restrict__ x,
                                                  float* __restrict__ mean,
                                                  float* __restrict__ invstd) {
    int c = blockIdx.x;          // 128 columns
    int tid = threadIdx.x;
    float s = 0.f, s2 = 0.f;
    for (int n = tid; n < NN; n += 256) {
        float v = x[(size_t)n * 128 + c];
        s += v; s2 += v * v;
    }
    __shared__ float sh[256], sh2[256];
    sh[tid] = s; sh2[tid] = s2;
    __syncthreads();
    for (int o = 128; o > 0; o >>= 1) {
        if (tid < o) { sh[tid] += sh[tid + o]; sh2[tid] += sh2[tid + o]; }
        __syncthreads();
    }
    if (tid == 0) {
        float mu  = sh[0] / (float)NN;
        float var = sh2[0] / (float)NN - mu * mu;
        mean[c] = mu;
        invstd[c] = rsqrtf(var + 1e-5f);
    }
}

__global__ void k_bn_apply(const float* __restrict__ x, const float* __restrict__ hres,
                           const float* __restrict__ g, const float* __restrict__ bt,
                           const float* __restrict__ mean, const float* __restrict__ invstd,
                           float* __restrict__ out, int domask) {
    int i = blockIdx.x * blockDim.x + threadIdx.x;
    if (i >= NN * 128) return;
    int c = i & 127;
    float v = g[c] * (x[i] - mean[c]) * invstd[c] + bt[c];
    v = v > 0.f ? v : (__expf(v) - 1.f);           // ELU
    float r = hres[i] + v;                          // residual
    if (domask && isinf(r)) r = 1e9f;
    out[i] = r;
}

// ---------------- bilevel pooling ----------------
__global__ void k_softmax100(const float* __restrict__ sraw, float* __restrict__ sout) {
    int n = blockIdx.x * blockDim.x + threadIdx.x;
    if (n >= NN) return;
    const float* p = sraw + (size_t)n * 112;
    float m = -1e30f;
    for (int a = 0; a < 100; ++a) m = fmaxf(m, p[a]);
    float sum = 0.f;
    for (int a = 0; a < 100; ++a) sum += __expf(p[a] - m);
    float inv = 1.f / sum;
    float* o = sout + (size_t)n * 100;
    for (int a = 0; a < 100; ++a) o[a] = __expf(p[a] - m) * inv;
}

__global__ __launch_bounds__(256) void k_coarse(const float* __restrict__ s,
                                                const float* __restrict__ hf,
                                                float* __restrict__ hc, int chunk) {
    __shared__ float acc[100 * 128];
    int tid = threadIdx.x;
    for (int i = tid; i < 100 * 128; i += 256) acc[i] = 0.f;
    __syncthreads();
    int n0 = blockIdx.x * chunk, n1 = n0 + chunk;
    for (int n = n0; n < n1; ++n) {
        const float* sr = s + (size_t)n * 100;
        const float* fr = hf + (size_t)n * 128;
        for (int i = tid; i < 100 * 128; i += 256) {
            int a = i >> 7, c = i & 127;
            acc[i] += sr[a] * fr[c];
        }
    }
    __syncthreads();
    for (int i = tid; i < 100 * 128; i += 256) atomicAdd(&hc[i], acc[i]);
}

__global__ void k_mix(const float* __restrict__ hfine, const float* __restrict__ s,
                      const float* __restrict__ hc, float* __restrict__ hout) {
    int i = blockIdx.x * blockDim.x + threadIdx.x;
    if (i >= NN * 128) return;
    int n = i >> 7, c = i & 127;
    const float* sr = s + (size_t)n * 100;
    float a = 0.f;
    for (int k = 0; k < 100; ++k) a += sr[k] * hc[k * 128 + c];
    float v = 0.5f * hfine[i] + 0.5f * a;           // SIGMA = 0.5
    if (isinf(v)) v = 1e9f;
    hout[i] = v;
}

// ---------------- fused edge MLP readout: 16 edges per block, no HBM temporaries --
// concat(h[src],h[dst]) tile (16x256 f16, 8KB) gathered global->LDS via async-DMA.
__global__ __launch_bounds__(256) void k_edge_mlp(const _Float16* __restrict__ hb,
                                                  const int* __restrict__ src,
                                                  const int* __restrict__ dst,
                                                  const _Float16* __restrict__ Wm1p,
                                                  const float* __restrict__ bm1,
                                                  const _Float16* __restrict__ Wm2p,
                                                  const float* __restrict__ bm2,
                                                  const float* __restrict__ Wm3,
                                                  const float* __restrict__ bm3,
                                                  float* __restrict__ zout) {
    __shared__ __align__(16) _Float16 atile[16 * 256];
    __shared__ __align__(16) _Float16 z1s[16 * 128];
    __shared__ _Float16 z2s[16 * 64];
    int e0 = blockIdx.x * 16;
    int tid = threadIdx.x;

    // stage 0: async gather of 16 concat rows (512 x 16B chunks, 2 per thread)
    for (int c = tid; c < 512; c += 256) {
        int m = c >> 5;                  // edge row 0..15
        int q = c & 31;                  // 16B chunk within 512B concat row
        int e = e0 + m;
        int node = (q < 16) ? src[e] : dst[e];
        async_ld16((unsigned)(uintptr_t)(atile + c * 8),
                   hb + (size_t)node * 128 + (q & 15) * 8);
    }
    wait_async0();
    __syncthreads();

    int wave = tid >> 5, lane = tid & 31;
    int n = lane & 15, mb = (lane >> 4) * 8;

    // stage 1: z1 = relu(concat @ Wm1 + bm1)  [16 x 128], K=256, nt=8
    {
        v8f acc = {};
#pragma unroll
        for (int kt = 0; kt < 8; ++kt) {
            v16h af = load_a_frag(atile, 256, kt * 32, lane);
            const v16h* bp = (const v16h*)(Wm1p + (size_t)((kt * 8 + wave) << 9));
            v16h bf = bp[lane];
            acc = __builtin_amdgcn_wmma_f32_16x16x32_f16(false, af, false, bf,
                                                         (short)0, acc, false, false);
        }
        int col0 = wave * 16;
        float bv = bm1[col0 + n];
#pragma unroll
        for (int r = 0; r < 8; ++r) {
            float v = acc[r] + bv;
            v = v > 0.f ? v : 0.f;
            z1s[(mb + r) * 128 + col0 + n] = (_Float16)v;
        }
    }
    __syncthreads();

    // stage 2: z2 = relu(z1 @ Wm2 + bm2)  [16 x 64], K=128, nt=4
    if (wave < 4) {
        v8f acc = {};
#pragma unroll
        for (int kt = 0; kt < 4; ++kt) {
            v16h af = load_a_frag(z1s, 128, kt * 32, lane);
            const v16h* bp = (const v16h*)(Wm2p + (size_t)((kt * 4 + wave) << 9));
            v16h bf = bp[lane];
            acc = __builtin_amdgcn_wmma_f32_16x16x32_f16(false, af, false, bf,
                                                         (short)0, acc, false, false);
        }
        int col0 = wave * 16;
        float bv = bm2[col0 + n];
#pragma unroll
        for (int r = 0; r < 8; ++r) {
            float v = acc[r] + bv;
            v = v > 0.f ? v : 0.f;
            z2s[(mb + r) * 64 + col0 + n] = (_Float16)v;
        }
    }
    __syncthreads();

    // stage 3: z = mask(z2 @ Wm3 + bm3)  [16 x 2]
    if (tid < 32) {
        int m = tid >> 1, j = tid & 1;
        float s = bm3[j];
        for (int k = 0; k < 64; ++k) s += (float)z2s[m * 64 + k] * Wm3[k * 2 + j];
        if (isinf(s)) s = 1e9f;
        zout[(size_t)(e0 + m) * 2 + j] = s;
    }
}

// ================================ host ========================================
extern "C" void kernel_launch(void* const* d_in, const int* in_sizes, int n_in,
                              void* d_out, int out_size, void* d_ws, size_t ws_size,
                              hipStream_t stream) {
    (void)in_sizes; (void)n_in; (void)out_size; (void)ws_size;
    const float* h0    = (const float*)d_in[0];
    const int*   src   = (const int*)d_in[2];
    const int*   dst   = (const int*)d_in[3];
    const float* W_emb = (const float*)d_in[4];
    const float* b_emb = (const float*)d_in[5];
    const float* Wg    = (const float*)d_in[6];
    const float* alg   = (const float*)d_in[7];
    const float* arg   = (const float*)d_in[8];
    const float* gam   = (const float*)d_in[10];
    const float* bet   = (const float*)d_in[11];
    const float* W3    = (const float*)d_in[12];
    const float* al3   = (const float*)d_in[13];
    const float* ar3   = (const float*)d_in[14];
    const float* gam3  = (const float*)d_in[16];
    const float* bet3  = (const float*)d_in[17];
    const float* Wa    = (const float*)d_in[18];
    const float* ba    = (const float*)d_in[19];
    const float* Wm1   = (const float*)d_in[20];
    const float* bm1   = (const float*)d_in[21];
    const float* Wm2   = (const float*)d_in[22];
    const float* bm2   = (const float*)d_in[23];
    const float* Wm3   = (const float*)d_in[24];
    const float* bm3   = (const float*)d_in[25];

    float* zout = (float*)d_out;                          // [E,2]
    float* sout = (float*)d_out + (size_t)EE * 2;         // [N,100]

    // ---- workspace carve ----
    char* w = (char*)d_ws;
    auto carve = [&](size_t bytes) { void* p = (void*)w; w += (bytes + 255) & ~(size_t)255; return p; };
    _Float16* hb     = (_Float16*)carve((size_t)NN * 128 * 2);
    float*    feat   = (float*)carve((size_t)NN * 128 * 4);
    float*    gout   = (float*)carve((size_t)NN * 128 * 4);
    float*    hbuf   = (float*)carve((size_t)NN * 128 * 4);
    float*    hfine  = (float*)carve((size_t)NN * 128 * 4);
    float*    el     = (float*)carve((size_t)NN * 8 * 4);
    float*    er     = (float*)carve((size_t)NN * 8 * 4);
    unsigned* mkey   = (unsigned*)carve((size_t)NN * 8 * 4);
    float*    den    = (float*)carve((size_t)NN * 8 * 4);
    float*    ex     = (float*)carve((size_t)EE * 8 * 4);
    float*    meanb  = (float*)carve(128 * 4);
    float*    invstb = (float*)carve(128 * 4);
    float*    hc     = (float*)carve(100 * 128 * 4);
    float*    sraw   = (float*)carve((size_t)NN * 112 * 4);
    _Float16* wWemb  = (_Float16*)carve(128 * 128 * 2);
    _Float16* wWg    = (_Float16*)carve(3 * 128 * 128 * 2);
    _Float16* wW3    = (_Float16*)carve(128 * 128 * 2);
    _Float16* wWa    = (_Float16*)carve(128 * 112 * 2);
    _Float16* wWm1   = (_Float16*)carve(256 * 128 * 2);
    _Float16* wWm2   = (_Float16*)carve(128 * 64 * 2);

    // ---- weight packing (f32 -> fragment-order f16) ----
    k_pack<<<GRID(128 * 128), 256, 0, stream>>>(W_emb, wWemb, 128, 128, 128);
    for (int l = 0; l < 3; ++l)
        k_pack<<<GRID(128 * 128), 256, 0, stream>>>(Wg + l * 128 * 128,
                                                    wWg + l * 128 * 128, 128, 128, 128);
    k_pack<<<GRID(128 * 128), 256, 0, stream>>>(W3, wW3, 128, 128, 128);
    k_pack<<<GRID(128 * 112), 256, 0, stream>>>(Wa, wWa, 128, 112, 100);
    k_pack<<<GRID(256 * 128), 256, 0, stream>>>(Wm1, wWm1, 256, 128, 128);
    k_pack<<<GRID(128 * 64), 256, 0, stream>>>(Wm2, wWm2, 128, 64, 64);

    // ---- embedding: h = h0 @ W_emb + b_emb ----
    k_f32_to_f16<<<GRID(NN * 128), 256, 0, stream>>>(h0, hb, NN * 128);
    k_gemm<<<NN / 16, 256, 0, stream>>>(hb, wWemb, b_emb, 128, hbuf, 128, 0);

    // ---- one GAT layer (fc -> attention -> edge softmax -> scatter -> BN/ELU/residual) --
    auto run_gat = [&](const _Float16* wf, const float* al, const float* ar,
                       const float* g, const float* bt, int H, int Dshift,
                       float* out_nodes, int domask) {
        int D = 1 << Dshift;
        k_f32_to_f16<<<GRID(NN * 128), 256, 0, stream>>>(hbuf, hb, NN * 128);
        k_gemm<<<NN / 16, 256, 0, stream>>>(hb, wf, nullptr, 0, feat, 128, 0);
        k_el_er<<<GRID(NN * H), 256, 0, stream>>>(feat, al, ar, el, er, H, D);
        hipMemsetAsync(mkey, 0, (size_t)NN * H * 4, stream);
        hipMemsetAsync(den, 0, (size_t)NN * H * 4, stream);
        hipMemsetAsync(gout, 0, (size_t)NN * 128 * 4, stream);
        k_edge_max<<<GRID(EE * H), 256, 0, stream>>>(el, er, src, dst, mkey, H);
        k_edge_exp<<<GRID(EE * H), 256, 0, stream>>>(el, er, src, dst, mkey, den, ex, H);
        k_scatter<<<(EE * 128) / 256, 256, 0, stream>>>(feat, ex, den, src, dst, gout, H, Dshift);
        k_bn_stats<<<128, 256, 0, stream>>>(gout, meanb, invstb);
        k_bn_apply<<<GRID(NN * 128), 256, 0, stream>>>(gout, hbuf, g, bt, meanb, invstb,
                                                       out_nodes, domask);
    };

    // layer 0
    run_gat(wWg, alg, arg, gam, bet, 8, 4, hbuf, 1);

    // layer 1 (biGAT): h_fine + soft assignment + coarse mixing
    run_gat(wWg + 128 * 128, alg + 128, arg + 128, gam + 128, bet + 128, 8, 4, hfine, 0);
    // hb still holds f16(h_in) from run_gat's conversion
    k_gemm<<<NN / 16, 256, 0, stream>>>(hb, wWa, ba, 100, sraw, 112, 0);
    k_softmax100<<<GRID(NN), 256, 0, stream>>>(sraw, sout);
    hipMemsetAsync(hc, 0, 100 * 128 * 4, stream);
    k_coarse<<<250, 256, 0, stream>>>(sout, hfine, hc, NN / 250);
    k_mix<<<GRID(NN * 128), 256, 0, stream>>>(hfine, sout, hc, hbuf);

    // layer 2
    run_gat(wWg + 2 * 128 * 128, alg + 256, arg + 256, gam + 256, bet + 256, 8, 4, hbuf, 1);
    // layer 3: 1 head, 128 dims
    run_gat(wW3, al3, ar3, gam3, bet3, 1, 7, hbuf, 1);

    // ---- fused edge MLP readout ----
    k_f32_to_f16<<<GRID(NN * 128), 256, 0, stream>>>(hbuf, hb, NN * 128);
    k_edge_mlp<<<EE / 16, 256, 0, stream>>>(hb, src, dst, wWm1, bm1, wWm2, bm2, Wm3, bm3, zout);
}
